// SpatialFrequencyLoss_4578435138099
// MI455X (gfx1250) — matrix operand order repaired and checked by reading, not compiled, not run on hardware
//
#include <hip/hip_runtime.h>
#include <hip/hip_bf16.h>
#include <math.h>

typedef __attribute__((ext_vector_type(16))) _Float16 v16h;
typedef __attribute__((ext_vector_type(8)))  _Float16 h8;
typedef __attribute__((ext_vector_type(8)))  float    v8f;

#define HW 512
#define NPLANES 12
#define NTOT (NPLANES * HW * HW)   // 3,145,728

// Per-sigma tables: k = odd(ceil(8*sigma)), p = k/2, p8 = roundup8(p),
// chunks = ceil((k + (p8-p) + 15)/32)
constexpr int   cK [6] = {5, 11, 21, 39, 77, 155};
constexpr int   cP [6] = {2, 5, 10, 19, 38, 77};
constexpr int   cP8[6] = {8, 8, 16, 24, 40, 80};
constexpr int   cCH[6] = {1, 1, 2, 2, 3, 6};
constexpr int   cCS[6] = {0, 1, 2, 4, 6, 9};   // prefix sum of cCH
constexpr float cSIG[6] = {0.6f, 1.2f, 2.4f, 4.8f, 9.6f, 19.2f};
constexpr float cW  [6] = {600.f, 500.f, 400.f, 20.f, 10.f, 10.f};

// ---------------------------------------------------------------------------
// Setup: build all banded B fragments (pass-H and pass-V, per sigma/branch/
// chunk) directly in WMMA wave32 B-matrix lane layout. 90 entries of
// 32x16 f16 = 512 halves each.
// ---------------------------------------------------------------------------
__global__ __launch_bounds__(256) void sfl_setupB(_Float16* __restrict__ Bm) {
  int bid  = blockIdx.x;           // [0,90)
  int pass = bid / 45;
  int rem  = bid % 45;
  int s = 0;
  while (s < 5 && rem >= 3 * cCS[s + 1]) ++s;
  int local = rem - 3 * cCS[s];
  int br = local / cCH[s];
  int ch = local % cCH[s];

  float sig = cSIG[s];
  int k = cK[s], p = cP[s], p8 = cP8[s];
  float ss2 = 2.f * sig * sig;
  float inv = 1.f / (2.f * 3.14159265358979323846f * sig * sig * sig * sig);

  // DC-correction constant c = sum(a_pre)/k^2 = 2*Sg*Su/k^2
  float Sg = 0.f, Su = 0.f;
  for (int t = -p; t <= p; ++t) {
    float tf = (float)t;
    float g = __expf(-(tf * tf) / ss2);
    Sg += g;
    Su += (tf * tf - sig * sig) * g * inv;
  }
  float c = 2.f * Sg * Su / ((float)k * (float)k);

  _Float16* ent = Bm + (size_t)(pass * 45 + 3 * cCS[s] + br * cCH[s] + ch) * 512;
  int shift = p8 - p;

  for (int h = threadIdx.x; h < 512; h += 256) {
    int lane  = h >> 4;           // which lane's slot
    int e     = h & 15;           // element within v16h
    int n     = lane & 15;        // B column
    int khalf = lane >> 4;        // 0: K 0-7,16-23, 1: K 8-15,24-31
    int j  = ch * 32 + ((e & 8) << 1) + khalf * 8 + (e & 7) - n;  // tap index in shifted kernel
    int jj = j - shift;
    float v = 0.f;
    if (jj >= 0 && jj < k) {
      float tf = (float)(jj - p);
      float g = __expf(-(tf * tf) / ss2);
      float u = (tf * tf - sig * sig) * g * inv;
      if (pass == 0) v = (br == 0) ? u : (br == 1) ? g : 1.f;      // H: u,g,box
      else           v = (br == 0) ? g : (br == 1) ? u : -c;       // V: g,u,-c
    }
    ent[h] = (_Float16)v;
  }
}

// ---------------------------------------------------------------------------
// diff = input - target, stored f16 (loss is linear in the conv input)
// ---------------------------------------------------------------------------
__global__ __launch_bounds__(256) void sfl_diff(const float* __restrict__ a,
                                                const float* __restrict__ b,
                                                _Float16* __restrict__ d, int n) {
  int i = blockIdx.x * 256 + threadIdx.x;
  if (i < n) d[i] = (_Float16)(a[i] - b[i]);
}

__device__ __forceinline__ int reflect512(int t) {
  t = (t < 0) ? -t : t;
  return (t > 511) ? (1022 - t) : t;
}

__device__ __forceinline__ v16h make_afrag(const _Float16* ap) {
  h8 lo = *(const h8*)(ap);        // K base..base+7  (ds_load_b128)
  h8 hi = *(const h8*)(ap + 16);   // K base+16..base+23
  v16h a;
#pragma unroll
  for (int e = 0; e < 8; ++e) { a[e] = lo[e]; a[8 + e] = hi[e]; }
  return a;
}

// Copy one 8-half block into LDS; vector 16B copy when source contiguous
// in-range, scalar reflect fallback at the borders.
__device__ __forceinline__ void stage_block(_Float16* __restrict__ dp,
                                            const _Float16* __restrict__ srow,
                                            int xg) {
  if (xg >= 0 && xg <= HW - 8) {
    *(h8*)dp = *(const h8*)(srow + xg);      // global b128 -> ds b128
  } else {
#pragma unroll
    for (int e = 0; e < 8; ++e) dp[e] = srow[reflect512(xg + e)];
  }
}

// ---------------------------------------------------------------------------
// Pass H: horizontal 1-D convs (u, g, box) via WMMA; writes 3 f16 planes
// TRANSPOSED ([x][y]) so pass V is again a contiguous row conv.
// grid = (32 row-blocks, 12 planes), block = 128 (4 waves)
// ---------------------------------------------------------------------------
template <int CHUNKS, int P8>
__global__ __launch_bounds__(128) void sfl_passH(
    const _Float16* __restrict__ diff,
    _Float16* __restrict__ pu, _Float16* __restrict__ pg, _Float16* __restrict__ pb,
    const _Float16* __restrict__ Bm, int bbase) {
  constexpr int L   = 496 + 32 * CHUNKS;   // staged halves per row (mult 8)
  constexpr int NB  = L / 8;
  constexpr int STR = 696;                 // 348 dwords == 28 mod 64 banks
  __shared__ _Float16 lds[16 * STR];

  const int plane = blockIdx.y;
  const int y0 = blockIdx.x * 16;
  const _Float16* src = diff + (size_t)plane * HW * HW;

  for (int i = threadIdx.x; i < 16 * NB; i += 128) {
    int r = i / NB, b8 = (i % NB) * 8;
    stage_block(lds + r * STR + b8, src + (y0 + r) * HW, b8 - P8);
  }
  __syncthreads();

  const int wave = threadIdx.x >> 5, lane = threadIdx.x & 31;
  const _Float16* arow = lds + (lane & 15) * STR + ((lane >> 4) << 3);
  _Float16* outp[3] = {pu, pg, pb};

#pragma unroll
  for (int b = 0; b < 3; ++b) {
    // hoist this branch's B fragments into registers (<= 6 x v16h)
    v16h Bf[CHUNKS];
    const v16h* bp = (const v16h*)(Bm + (size_t)(bbase + b * CHUNKS * 512));
#pragma unroll
    for (int ch = 0; ch < CHUNKS; ++ch) Bf[ch] = bp[ch * 32 + lane];

    for (int tx = wave; tx < 32; tx += 4) {
      const int x0 = tx * 16;
      v8f acc = {};
#pragma unroll
      for (int ch = 0; ch < CHUNKS; ++ch)
        acc = __builtin_amdgcn_wmma_f32_16x16x32_f16(
            false, make_afrag(arow + x0 + ch * 32), false, Bf[ch],
            (short)0, acc, false, false);
      // transposed store: lane owns column x = x0 + (lane&15); rows
      // y = y0 + r + 8*(lane>>4) are contiguous -> one 16B store
      h8 o;
#pragma unroll
      for (int e = 0; e < 8; ++e) o[e] = (_Float16)acc[e];
      _Float16* dst = outp[b] + (size_t)plane * HW * HW +
                      (size_t)(x0 + (lane & 15)) * HW + y0 + ((lane >> 4) << 3);
      *(h8*)dst = o;
    }
  }
}

// ---------------------------------------------------------------------------
// Pass V: vertical convs on the transposed planes, fused over the 3 branches
// into one accumulator; square, wave-reduce, atomic add of w/NCHW * sum.
// grid = (32 x-blocks, 12 planes, 2 y-halves), block = 128
// ---------------------------------------------------------------------------
template <int CHUNKS, int P8>
__global__ __launch_bounds__(128) void sfl_passV(
    const _Float16* __restrict__ pu, const _Float16* __restrict__ pg,
    const _Float16* __restrict__ pb, float* __restrict__ out,
    const _Float16* __restrict__ Bm, int bbase, float scale) {
  constexpr int L   = 240 + 32 * CHUNKS;   // staged halves per row (mult 8)
  constexpr int NB  = L / 8;
  constexpr int STR = 440;                 // 220 dwords == 28 mod 64 banks
  __shared__ _Float16 lds[3 * 16 * STR];

  const int plane = blockIdx.y;
  const int x0 = blockIdx.x * 16;
  const int ybase = blockIdx.z * 256;

  for (int i = threadIdx.x; i < 3 * 16 * NB; i += 128) {
    int b = i / (16 * NB);
    int rr = i % (16 * NB);
    int r = rr / NB, b8 = (rr % NB) * 8;
    const _Float16* srow = (b == 0 ? pu : b == 1 ? pg : pb) +
                           (size_t)plane * HW * HW + (size_t)(x0 + r) * HW;
    stage_block(lds + b * 16 * STR + r * STR + b8, srow, ybase + b8 - P8);
  }
  __syncthreads();

  const int wave = threadIdx.x >> 5, lane = threadIdx.x & 31;
  float ss = 0.f;

  for (int ty = wave; ty < 16; ty += 4) {
    const int yo = ty * 16;
    v8f acc = {};
#pragma unroll
    for (int b = 0; b < 3; ++b) {
      const v16h* bp = (const v16h*)(Bm + (size_t)(bbase + b * CHUNKS * 512));
      const _Float16* arow =
          lds + b * 16 * STR + (lane & 15) * STR + yo + ((lane >> 4) << 3);
#pragma unroll
      for (int ch = 0; ch < CHUNKS; ++ch)
        acc = __builtin_amdgcn_wmma_f32_16x16x32_f16(
            false, make_afrag(arow + ch * 32), false, bp[ch * 32 + lane],
            (short)0, acc, false, false);
    }
#pragma unroll
    for (int e = 0; e < 8; ++e) ss += acc[e] * acc[e];
  }

  // wave32 reduction + one atomic per wave
#pragma unroll
  for (int off = 16; off > 0; off >>= 1) ss += __shfl_down(ss, off, 32);
  if (lane == 0) atomicAdd(out, ss * scale);
}

// ---------------------------------------------------------------------------
// Workspace layout (bytes):
//   [0, 92160)            : 90 B-fragments (f16, 512 halves each)
//   [131072, +6291456)    : diff plane (f16, 12x512x512)
//   then 3 intermediate planes (f16) -> total ~24.1 MB
// ---------------------------------------------------------------------------
extern "C" void kernel_launch(void* const* d_in, const int* in_sizes, int n_in,
                              void* d_out, int out_size, void* d_ws, size_t ws_size,
                              hipStream_t stream) {
  const float* in = (const float*)d_in[0];
  const float* tg = (const float*)d_in[1];
  float* out = (float*)d_out;
  char* ws = (char*)d_ws;

  const size_t PLANE_B = (size_t)NTOT * sizeof(_Float16);   // 6,291,456
  _Float16* Bm    = (_Float16*)(ws);
  _Float16* diffp = (_Float16*)(ws + 131072);
  _Float16* pu    = (_Float16*)(ws + 131072 + 1 * PLANE_B);
  _Float16* pg    = (_Float16*)(ws + 131072 + 2 * PLANE_B);
  _Float16* pb    = (_Float16*)(ws + 131072 + 3 * PLANE_B);

  hipMemsetAsync(d_out, 0, sizeof(float), stream);
  sfl_setupB<<<90, 256, 0, stream>>>(Bm);
  sfl_diff<<<(NTOT + 255) / 256, 256, 0, stream>>>(in, tg, diffp, NTOT);

  const dim3 gH(32, NPLANES), gV(32, NPLANES, 2);
  for (int s = 0; s < 6; ++s) {
    const int bbaseH = 3 * cCS[s] * 512;
    const int bbaseV = (45 + 3 * cCS[s]) * 512;
    const float sc = cW[s] / (float)NTOT;
    switch (s) {
      case 0:
      case 1:
        sfl_passH<1, 8><<<gH, 128, 0, stream>>>(diffp, pu, pg, pb, Bm, bbaseH);
        sfl_passV<1, 8><<<gV, 128, 0, stream>>>(pu, pg, pb, out, Bm, bbaseV, sc);
        break;
      case 2:
        sfl_passH<2, 16><<<gH, 128, 0, stream>>>(diffp, pu, pg, pb, Bm, bbaseH);
        sfl_passV<2, 16><<<gV, 128, 0, stream>>>(pu, pg, pb, out, Bm, bbaseV, sc);
        break;
      case 3:
        sfl_passH<2, 24><<<gH, 128, 0, stream>>>(diffp, pu, pg, pb, Bm, bbaseH);
        sfl_passV<2, 24><<<gV, 128, 0, stream>>>(pu, pg, pb, out, Bm, bbaseV, sc);
        break;
      case 4:
        sfl_passH<3, 40><<<gH, 128, 0, stream>>>(diffp, pu, pg, pb, Bm, bbaseH);
        sfl_passV<3, 40><<<gV, 128, 0, stream>>>(pu, pg, pb, out, Bm, bbaseV, sc);
        break;
      default:
        sfl_passH<6, 80><<<gH, 128, 0, stream>>>(diffp, pu, pg, pb, Bm, bbaseH);
        sfl_passV<6, 80><<<gV, 128, 0, stream>>>(pu, pg, pb, out, Bm, bbaseV, sc);
        break;
    }
  }
}
